// DeepRecurrentModel_4776003633717
// MI455X (gfx1250) — matrix operand order repaired and checked by reading, not compiled, not run on hardware
//
#include <hip/hip_runtime.h>
#include <stdint.h>

// ---------------- problem constants ----------------
constexpr int kB = 256, kS = 1024, kI = 64, kH = 256, kO = 32;
constexpr size_t GI_ELEMS = (size_t)kS * kB * 3 * kH;   // 1024*256*768 f32
constexpr size_t GI_BYTES = GI_ELEMS * 4;               // 768 MB

// bf16-packed weight offsets (elements) inside ws after the gi region
constexpr int OFF_EW0 = 0;                       // enc_w0 [256][64]
constexpr int OFF_EW1 = OFF_EW0 + kH * kI;       // enc_w1 [256][256]
constexpr int OFF_EW2 = OFF_EW1 + kH * kH;       // enc_w2 [256][256]
constexpr int OFF_WIH = OFF_EW2 + kH * kH;       // gru_wih [768][256]
constexpr int OFF_WHH = OFF_WIH + 3 * kH * kH;   // gru_whh [768][256]
constexpr int OFF_DW0 = OFF_WHH + 3 * kH * kH;   // dec_w0 [256][256]
constexpr int OFF_DW1 = OFF_DW0 + kH * kH;       // dec_w1 [256][256]
constexpr int OFF_DW2 = OFF_DW1 + kH * kH;       // dec_w2 [32][256]

// ---------------- WMMA fragment types ----------------
typedef __attribute__((ext_vector_type(16))) __bf16 bf16x16;
typedef __attribute__((ext_vector_type(8)))  float  f32x8;

union FragAB { bf16x16 v; uint32_t u[8]; };
union FragC  { f32x8 v; float f[8]; };

// Native f32 -> bf16 (lets the compiler use v_cvt_*_bf16_f32 instead of a
// software RNE bit sequence).
__device__ __forceinline__ unsigned short f2bf(float f) {
  union { __bf16 h; unsigned short u; } c;
  c.h = (__bf16)f;
  return c.u;
}

__device__ __forceinline__ f32x8 wmma_bf16(bf16x16 a, bf16x16 b, f32x8 c) {
  return __builtin_amdgcn_wmma_f32_16x16x32_bf16(
      /*neg_a=*/false, a, /*neg_b=*/false, b,
      /*c_mod=*/(short)0, c, /*reuse_a=*/false, /*reuse_b=*/false);
}

// ---- CDNA5 async global->LDS copy (ASYNCcnt-tracked) ----
// Builtin signature (from hipcc diagnostic): param0 is v4i* in the global
// address space, param1 is v4i* in LDS, then imm offset + cpol.
typedef __attribute__((ext_vector_type(4))) int v4i;
typedef __attribute__((address_space(1))) v4i* as1_v4i;
typedef __attribute__((address_space(3))) v4i* as3_v4i;

__device__ __forceinline__ void async_load_b128(const void* g, void* l) {
#if __has_builtin(__builtin_amdgcn_global_load_async_to_lds_b128)
  __builtin_amdgcn_global_load_async_to_lds_b128(
      (as1_v4i)const_cast<void*>(g), (as3_v4i)l, 0, 0);
#else
  uint32_t loff = (uint32_t)(uintptr_t)(as3_v4i)l;
  asm volatile("global_load_async_to_lds_b128 %0, %1, off"
               :: "v"(loff), "v"(g) : "memory");
#endif
}

#if __has_builtin(__builtin_amdgcn_s_wait_asynccnt)
#define WAIT_ASYNC(n) __builtin_amdgcn_s_wait_asynccnt(n)
#else
#define WAIT_ASYNC(n) asm volatile("s_wait_asynccnt " #n ::: "memory")
#endif

// A fragment: 16x32 bf16 tile. s points at element (m0, k0) of an LDS bf16
// matrix with row stride `stride` (elements, even). ISA layout: lane = row,
// half-wave selects K sub-block: lanes 0-15 hold K = {0..7,16..23},
// lanes 16-31 hold K = {8..15,24..31}; VGPR v packs K pair (lo=even K).
__device__ __forceinline__ FragAB load_a(const unsigned short* s, int stride, int lane) {
  FragAB a;
  int m    = lane & 15;
  int koff = (lane >> 4) << 3;                   // 0 or 8
  const uint32_t* p0 = (const uint32_t*)(s + m * stride + koff);
  const uint32_t* p1 = (const uint32_t*)(s + m * stride + 16 + koff);
  a.u[0] = p0[0]; a.u[1] = p0[1]; a.u[2] = p0[2]; a.u[3] = p0[3];
  a.u[4] = p1[0]; a.u[5] = p1[1]; a.u[6] = p1[2]; a.u[7] = p1[3];
  return a;
}

// B fragment: 32x16 tile of B where B[k][n] = W[n][k], W row-major [N][Kdim]
// bf16 in global. Lane = column n, half-wave selects K sub-block: lanes 0-15
// hold K = {0..7,16..23}, lanes 16-31 hold K = {8..15,24..31} (matches load_a).
__device__ __forceinline__ FragAB load_b(const unsigned short* W, int Kdim,
                                         int n0, int k0, int lane) {
  FragAB b;
  int n  = n0 + (lane & 15);
  int kb = k0 + ((lane >> 4) << 3);
  const uint32_t* q = (const uint32_t*)(W + (size_t)n * Kdim + kb);
#pragma unroll
  for (int v = 0; v < 8; ++v) b.u[v] = q[(v & 3) + ((v >> 2) << 3)];
  return b;
}

// out = opt_relu(act @ W^T + bias): act bf16 LDS [M][strideIn], W bf16 global
// [N][K], result bf16 -> LDS [M][strideOut]. Tiles distributed across waves.
__device__ __forceinline__ void mlp_stage(const unsigned short* sIn, int strideIn,
                                          unsigned short* sOut, int strideOut,
                                          const unsigned short* W, const float* bias,
                                          int M, int N, int K, bool relu,
                                          int wave, int nWaves, int lane) {
  const int mTiles = M >> 4, nTiles = N >> 4, kSteps = K >> 5;
  for (int tile = wave; tile < mTiles * nTiles; tile += nWaves) {
    int mt = tile / nTiles, nt = tile % nTiles;
    FragC acc;
#pragma unroll
    for (int r = 0; r < 8; ++r) acc.f[r] = 0.f;
    for (int ks = 0; ks < kSteps; ++ks) {
      FragAB a = load_a(sIn + (mt * 16) * strideIn + ks * 32, strideIn, lane);
      FragAB b = load_b(W, K, nt * 16, ks * 32, lane);
      acc.v = wmma_bf16(a.v, b.v, acc.v);
    }
    int col = nt * 16 + (lane & 15);
    float bb = bias[col];
    int r0 = (lane >> 4) * 8;
#pragma unroll
    for (int r = 0; r < 8; ++r) {
      float v = acc.f[r] + bb;
      if (relu) v = fmaxf(v, 0.f);
      sOut[(mt * 16 + r0 + r) * strideOut + col] = f2bf(v);
    }
  }
}

// ---------------- kernel 0: pack f32 weights -> bf16 ----------------
__global__ void pack_bf16_kernel(const float* __restrict__ src,
                                 unsigned short* __restrict__ dst, int n) {
  for (int i = blockIdx.x * blockDim.x + threadIdx.x; i < n;
       i += gridDim.x * blockDim.x)
    dst[i] = f2bf(src[i]);
}

// ---------------- kernel 1: fused normalize + encoder + gi ----------------
// 64 rows/block through x -> h1 -> h2 -> e -> gi (writes gi[s][b][768] f32).
__global__ __launch_bounds__(256) void encoder_gi_kernel(
    const float* __restrict__ x, const float* __restrict__ mean,
    const float* __restrict__ stdv, const unsigned short* __restrict__ wpack,
    const float* __restrict__ eb0, const float* __restrict__ eb1,
    const float* __restrict__ eb2, const float* __restrict__ bih,
    float* __restrict__ gi) {
  __shared__ unsigned short P[64 * 264];
  __shared__ unsigned short Q[64 * 264];
  const int tid = threadIdx.x, lane = tid & 31, wave = tid >> 5;
  const int row0 = blockIdx.x * 64;

  // normalize x tile -> bf16 in P (stride 72)
  for (int i = tid; i < 64 * 64; i += 256) {
    int r = i >> 6, c = i & 63;
    float v = (x[(size_t)(row0 + r) * kI + c] - mean[c]) / stdv[c];
    P[r * 72 + c] = f2bf(v);
  }
  __syncthreads();
  mlp_stage(P, 72, Q, 264, wpack + OFF_EW0, eb0, 64, kH, kI, true, wave, 8, lane);
  __syncthreads();
  mlp_stage(Q, 264, P, 264, wpack + OFF_EW1, eb1, 64, kH, kH, true, wave, 8, lane);
  __syncthreads();
  mlp_stage(P, 264, Q, 264, wpack + OFF_EW2, eb2, 64, kH, kH, false, wave, 8, lane);
  __syncthreads();

  // gi = e @ wih.T + bih : 64 x 768, K=256, f32 to global [s][b][768]
  const unsigned short* Wg = wpack + OFF_WIH;
  for (int tile = wave; tile < 4 * 48; tile += 8) {
    int mt = tile / 48, nt = tile % 48;
    FragC acc;
#pragma unroll
    for (int r = 0; r < 8; ++r) acc.f[r] = 0.f;
#pragma unroll
    for (int ks = 0; ks < 8; ++ks) {
      FragAB a = load_a(Q + (mt * 16) * 264 + ks * 32, 264, lane);
      FragAB b = load_b(Wg, kH, nt * 16, ks * 32, lane);
      acc.v = wmma_bf16(a.v, b.v, acc.v);
    }
    int col = nt * 16 + (lane & 15);
    float bb = bih[col];
    int r0 = (lane >> 4) * 8;
#pragma unroll
    for (int r = 0; r < 8; ++r) {
      int grow = row0 + mt * 16 + r0 + r;       // = b*1024 + s
      int bI = grow >> 10, sI = grow & 1023;
      gi[((size_t)sI * kB + bI) * (3 * kH) + col] = acc.f[r] + bb;
    }
  }
}

// ---------------- kernel 2: GRU recurrence ----------------
// 16 blocks x 512 threads (16 waves). Block owns 16 batch rows; h slice lives
// in LDS (f32 + bf16 shadow). Each wave owns one 16-column block of h and
// keeps all 24 whh B-fragments (r/z/n x 8 ksteps) resident in VGPRs.
// gi[t] slices (48 KB/step) are double-buffered in LDS via async global->LDS
// loads overlapped with the WMMA work of the current step.
__global__ __launch_bounds__(512) void gru_kernel(
    const float* __restrict__ carry, const unsigned short* __restrict__ wpack,
    const float* __restrict__ bhh, const float* __restrict__ gi,
    float* __restrict__ hT) {
  __shared__ float          hf[16 * 256];
  __shared__ unsigned short hb[16 * 264];
  __shared__ __align__(16) float sgi[2 * 16 * 768];   // double-buffered gi[t]
  const int tid = threadIdx.x, lane = tid & 31, wave = tid >> 5;
  const int b0 = blockIdx.x * 16;

  for (int i = tid; i < 16 * 256; i += 512) {
    int r = i >> 8, c = i & 255;
    float v = carry[(size_t)(b0 + r) * kH + c];
    hf[r * 256 + c] = v;
    hb[r * 264 + c] = f2bf(v);
  }

  const unsigned short* Whh = wpack + OFF_WHH;   // [768][256] bf16
  const int n0 = wave * 16;
  FragAB Br[8], Bz[8], Bn[8];
#pragma unroll
  for (int ks = 0; ks < 8; ++ks) {
    Br[ks] = load_b(Whh, kH, n0,          ks * 32, lane);
    Bz[ks] = load_b(Whh, kH, n0 + kH,     ks * 32, lane);
    Bn[ks] = load_b(Whh, kH, n0 + 2 * kH, ks * 32, lane);
  }
  const int col = n0 + (lane & 15);
  const float bhr = bhh[col], bhz = bhh[col + kH], bhn = bhh[col + 2 * kH];
  const int r0 = (lane >> 4) * 8;

  // preload gi[0] into buffer 0 (6 async b128 instructions per wave)
  {
    const float* src = gi + ((size_t)0 * kB + b0) * (3 * kH);
    for (int c = tid; c < 3072; c += 512)
      async_load_b128(src + c * 4, sgi + c * 4);
  }
  __syncthreads();

  float hnew[8];
  for (int t = 0; t < kS; ++t) {
    const float* cur = sgi + (t & 1) * (16 * 768);
    // kick off async copy of gi[t+1] into the other buffer
    if (t + 1 < kS) {
      const float* src = gi + ((size_t)(t + 1) * kB + b0) * (3 * kH);
      float* dst = sgi + ((t + 1) & 1) * (16 * 768);
      for (int c = tid; c < 3072; c += 512)
        async_load_b128(src + c * 4, dst + c * 4);
    }
    // gh tiles for the three gates (overlaps with the async copy above)
    FragC ar, az, an;
#pragma unroll
    for (int r = 0; r < 8; ++r) { ar.f[r] = 0.f; az.f[r] = 0.f; an.f[r] = 0.f; }
#pragma unroll
    for (int ks = 0; ks < 8; ++ks) {
      FragAB a = load_a(hb + ks * 32, 264, lane);
      ar.v = wmma_bf16(a.v, Br[ks].v, ar.v);
      az.v = wmma_bf16(a.v, Bz[ks].v, az.v);
      an.v = wmma_bf16(a.v, Bn[ks].v, an.v);
    }
    // drain this step's async loads (t+1's 6 may stay in flight), then make
    // the whole buffer visible workgroup-wide
    if (t + 1 < kS) { WAIT_ASYNC(6); } else { WAIT_ASYNC(0); }
    __syncthreads();
#pragma unroll
    for (int r = 0; r < 8; ++r) {
      int lrow = r0 + r;
      const float* grow = cur + lrow * (3 * kH);
      float gr = grow[col], gz = grow[col + kH], gn = grow[col + 2 * kH];
      float rg = 1.f / (1.f + __expf(-(gr + ar.f[r] + bhr)));
      float zg = 1.f / (1.f + __expf(-(gz + az.f[r] + bhz)));
      float ng = tanhf(gn + rg * (an.f[r] + bhn));
      float ho = hf[lrow * 256 + col];
      hnew[r] = (1.f - zg) * ng + zg * ho;
    }
    __syncthreads();
#pragma unroll
    for (int r = 0; r < 8; ++r) {
      int lrow = r0 + r;
      hf[lrow * 256 + col] = hnew[r];
      hb[lrow * 264 + col] = f2bf(hnew[r]);
    }
    __syncthreads();
  }
#pragma unroll
  for (int r = 0; r < 8; ++r) {
    int lrow = r0 + r;
    hT[(size_t)(b0 + lrow) * kH + col] = hf[lrow * 256 + col];
  }
}

// ---------------- kernel 3: decoder ----------------
__global__ __launch_bounds__(256) void decoder_kernel(
    const float* __restrict__ hT, const unsigned short* __restrict__ wpack,
    const float* __restrict__ db0, const float* __restrict__ db1,
    const float* __restrict__ db2, float* __restrict__ out) {
  __shared__ unsigned short P[64 * 264];
  __shared__ unsigned short Q[64 * 264];
  const int tid = threadIdx.x, lane = tid & 31, wave = tid >> 5;
  const int row0 = blockIdx.x * 64;

  for (int i = tid; i < 64 * 256; i += 256) {
    int r = i >> 8, c = i & 255;
    P[r * 264 + c] = f2bf(hT[(size_t)(row0 + r) * kH + c]);
  }
  __syncthreads();
  mlp_stage(P, 264, Q, 264, wpack + OFF_DW0, db0, 64, kH, kH, true, wave, 8, lane);
  __syncthreads();
  mlp_stage(Q, 264, P, 264, wpack + OFF_DW1, db1, 64, kH, kH, true, wave, 8, lane);
  __syncthreads();

  // final 64x32, K=256: 8 tiles over 8 waves, f32 to d_out
  const unsigned short* W = wpack + OFF_DW2;   // [32][256]
  int mt = wave >> 1, nt = wave & 1;
  FragC acc;
#pragma unroll
  for (int r = 0; r < 8; ++r) acc.f[r] = 0.f;
#pragma unroll
  for (int ks = 0; ks < 8; ++ks) {
    FragAB a = load_a(P + (mt * 16) * 264 + ks * 32, 264, lane);
    FragAB b = load_b(W, kH, nt * 16, ks * 32, lane);
    acc.v = wmma_bf16(a.v, b.v, acc.v);
  }
  int col = nt * 16 + (lane & 15);
  float bb = db2[col];
  int r0 = (lane >> 4) * 8;
#pragma unroll
  for (int r = 0; r < 8; ++r)
    out[(size_t)(row0 + mt * 16 + r0 + r) * kO + col] = acc.f[r] + bb;
}

// ---------------- host side ----------------
extern "C" void kernel_launch(void* const* d_in, const int* in_sizes, int n_in,
                              void* d_out, int out_size, void* d_ws, size_t ws_size,
                              hipStream_t stream) {
  const float* x      = (const float*)d_in[0];
  const float* carry  = (const float*)d_in[1];
  const float* mean   = (const float*)d_in[2];
  const float* stdv   = (const float*)d_in[3];
  const float* enc_w0 = (const float*)d_in[4];
  const float* enc_b0 = (const float*)d_in[5];
  const float* enc_w1 = (const float*)d_in[6];
  const float* enc_b1 = (const float*)d_in[7];
  const float* enc_w2 = (const float*)d_in[8];
  const float* enc_b2 = (const float*)d_in[9];
  const float* gru_wih = (const float*)d_in[10];
  const float* gru_whh = (const float*)d_in[11];
  const float* gru_bih = (const float*)d_in[12];
  const float* gru_bhh = (const float*)d_in[13];
  const float* dec_w0 = (const float*)d_in[14];
  const float* dec_b0 = (const float*)d_in[15];
  const float* dec_w1 = (const float*)d_in[16];
  const float* dec_b1 = (const float*)d_in[17];
  const float* dec_w2 = (const float*)d_in[18];
  const float* dec_b2 = (const float*)d_in[19];

  float* gi = (float*)d_ws;
  unsigned short* wpack = (unsigned short*)((char*)d_ws + GI_BYTES);

  auto pack = [&](const float* src, int off, int n) {
    int grid = (n + 255) / 256; if (grid > 512) grid = 512;
    pack_bf16_kernel<<<grid, 256, 0, stream>>>(src, wpack + off, n);
  };
  pack(enc_w0,  OFF_EW0, kH * kI);
  pack(enc_w1,  OFF_EW1, kH * kH);
  pack(enc_w2,  OFF_EW2, kH * kH);
  pack(gru_wih, OFF_WIH, 3 * kH * kH);
  pack(gru_whh, OFF_WHH, 3 * kH * kH);
  pack(dec_w0,  OFF_DW0, kH * kH);
  pack(dec_w1,  OFF_DW1, kH * kH);
  pack(dec_w2,  OFF_DW2, kO * kH);

  encoder_gi_kernel<<<(kB * kS) / 64, 256, 0, stream>>>(
      x, mean, stdv, wpack, enc_b0, enc_b1, enc_b2, gru_bih, gi);

  float* out = (float*)d_out;           // [256][32]
  float* hT  = out + kB * kO;           // [256][256]
  gru_kernel<<<kB / 16, 512, 0, stream>>>(carry, wpack, gru_bhh, gi, hT);
  decoder_kernel<<<kB / 64, 256, 0, stream>>>(hT, wpack, dec_b0, dec_b1, dec_b2, out);
}
